// EdgeConv_11811160064041
// MI455X (gfx1250) — matrix operand order repaired and checked by reading, not compiled, not run on hardware
//
#include <hip/hip_runtime.h>

typedef __attribute__((ext_vector_type(16))) __bf16  v16bf;
typedef __attribute__((ext_vector_type(8)))  float   v8f;
typedef __attribute__((ext_vector_type(4)))  float   v4f;
typedef __attribute__((ext_vector_type(4)))  unsigned int v4u;
typedef __attribute__((ext_vector_type(2)))  unsigned int v2u;

union FragU { v4u q[2]; v16bf v; };

static __device__ __forceinline__ unsigned short f32_to_bf16(float f) {
    unsigned int u = __builtin_bit_cast(unsigned int, f);
    u += 0x7FFFu + ((u >> 16) & 1u);      // round-to-nearest-even
    return (unsigned short)(u >> 16);
}

// ---------------------------------------------------------------------------
// Kernel 0: zero the accumulator (d_out doubles as msg_sum) and cnt[]
// ---------------------------------------------------------------------------
__global__ void init_zero(float* __restrict__ out, float* __restrict__ cnt,
                          long nOut, long nCnt) {
    long i = (long)blockIdx.x * blockDim.x + threadIdx.x;
    long stride = (long)gridDim.x * blockDim.x;
    for (long k = i; k < nOut; k += stride) out[k] = 0.f;
    for (long k = i; k < nCnt; k += stride) cnt[k] = 0.f;
}

// ---------------------------------------------------------------------------
// Kernel 1/2: pack W (128x64 f32 row-major) into bf16 WMMA B-fragments.
// Fragment (t,n): 32 lanes x 16 bf16 contiguous (lane-major).
//   lane < 16 : col = 16n+lane,    K = 32t + (j<8 ? j : 16+j-8)      (j = 0..15)
//   lane >= 16: col = 16n+lane-16, K = 32t + 8 + (j<8 ? j : 16+j-8)
// Pairs (j even, j odd) land in VGPR j/2 low/high halves — the ISA B layout.
// ---------------------------------------------------------------------------
__global__ void pack_w(const float* __restrict__ W, unsigned short* __restrict__ dst) {
    int o    = blockIdx.x * 256 + threadIdx.x;   // 0..8191
    int frag = o >> 9;
    int rem  = o & 511;
    int lane = rem >> 4;
    int j    = rem & 15;
    int t = frag >> 2, n = frag & 3;
    int c = (n << 4) + (lane & 15);
    int k = (t << 5) + ((lane >= 16) ? 8 : 0) + ((j & 8) << 1) + (j & 7);
    dst[o] = f32_to_bf16(W[k * 64 + c]);
}

// ---------------------------------------------------------------------------
// Shared GEMM core: A tile (16x128 bf16) lives in LDS row-major (stride 128),
// B fragments pre-packed in global ws. 16x v_wmma_f32_16x16x32_bf16.
// ---------------------------------------------------------------------------
static __device__ __forceinline__ void wmma_tile(
    const unsigned short* __restrict__ ldsRowBase,  // row (lane&15) of this wave's tile
    const unsigned short* __restrict__ wPacked,
    int lane, v8f acc[4]) {
    const int k0 = (lane >= 16) ? 8 : 0;            // A K-offset for this half
    #pragma unroll
    for (int t = 0; t < 4; ++t) {
        FragU a;
        const v4u* ap = (const v4u*)(ldsRowBase + t * 32 + k0);
        a.q[0] = ap[0];            // K = k0 .. k0+7
        a.q[1] = ap[2];            // K = k0+16 .. k0+23
        #pragma unroll
        for (int n = 0; n < 4; ++n) {
            FragU b;
            const v4u* bp = (const v4u*)(wPacked + (size_t)(t * 4 + n) * 512 + lane * 16);
            b.q[0] = bp[0];
            b.q[1] = bp[1];
            acc[n] = __builtin_amdgcn_wmma_f32_16x16x32_bf16(
                false, a.v, false, b.v, (short)0, acc[n], false, false);
        }
    }
}

// ---------------------------------------------------------------------------
// Kernel 3: edge MLP + scatter.  block = 256 thr = 8 waves, 128 edges/block.
// ---------------------------------------------------------------------------
__global__ __launch_bounds__(256) void edge_kernel(
    const float* __restrict__ hid, const float* __restrict__ ef,
    const int* __restrict__ src, const int* __restrict__ dst,
    const unsigned short* __restrict__ w1p, const float* __restrict__ b1,
    float* __restrict__ out, float* __restrict__ cnt, int E) {
    __shared__ unsigned short lds[128 * 128];   // 32 KB: [128 edges][128 = ef|hid[src]]

    const int tid = threadIdx.x;
    const long base = (long)blockIdx.x * 128;

    // Stage edge_in as bf16: 2 threads per edge row (half 0: edge_feat, half 1: hid gather)
    {
        const int  erow = tid >> 1;
        const int  half = tid & 1;
        const long e    = base + erow;
        const bool ok   = e < E;
        const float* srcp;
        if (half == 0) srcp = ef + (ok ? e : 0) * 64;
        else           srcp = hid + (long)(ok ? src[e] : 0) * 64;
        const float scale = ok ? 1.f : 0.f;
        unsigned short* lrow = lds + erow * 128 + half * 64;
        #pragma unroll
        for (int i = 0; i < 16; ++i) {
            v4f f = *(const v4f*)(srcp + 4 * i) * scale;
            v2u d;
            d.x = (unsigned)f32_to_bf16(f.x) | ((unsigned)f32_to_bf16(f.y) << 16);
            d.y = (unsigned)f32_to_bf16(f.z) | ((unsigned)f32_to_bf16(f.w) << 16);
            *(v2u*)(lrow + 8 * i) = d;
        }
    }
    __syncthreads();

    const int lane = tid & 31, wave = tid >> 5;
    v8f acc[4] = {v8f{}, v8f{}, v8f{}, v8f{}};
    wmma_tile(lds + (wave * 16 + (lane & 15)) * 128, w1p, lane, acc);

    // Epilogue: bias + ReLU + atomic scatter-add into out[dst]
    const int  mbase = (lane >= 16) ? 8 : 0;
    const long ebase = base + wave * 16;
    int  di[8]; bool ev[8];
    #pragma unroll
    for (int r = 0; r < 8; ++r) {
        long e = ebase + mbase + r;
        ev[r] = e < E;
        di[r] = ev[r] ? dst[e] : 0;
    }
    #pragma unroll
    for (int n = 0; n < 4; ++n) {
        const int col = (n << 4) + (lane & 15);
        const float bias = b1[col];
        #pragma unroll
        for (int r = 0; r < 8; ++r) {
            float v = acc[n][r] + bias;
            v = v > 0.f ? v : 0.f;
            if (ev[r]) atomicAdd(out + (long)di[r] * 64 + col, v);
        }
    }
    if (lane < 16) {            // one count per edge
        long e = ebase + lane;
        if (e < E) atomicAdd(cnt + dst[e], 1.0f);
    }
}

// ---------------------------------------------------------------------------
// Kernel 4: node MLP.  reduced = out/max(cnt,1); out = relu([reduced|hid]@W2+b2)
// ---------------------------------------------------------------------------
__global__ __launch_bounds__(256) void node_kernel(
    const float* __restrict__ hid, const unsigned short* __restrict__ w2p,
    const float* __restrict__ b2, const float* __restrict__ cnt,
    float* __restrict__ out, int N) {
    __shared__ unsigned short lds[128 * 128];

    const int tid = threadIdx.x;
    const long base = (long)blockIdx.x * 128;

    {
        const int  row  = tid >> 1;
        const int  half = tid & 1;
        const long node = base + row;
        const bool ok   = node < N;
        const float* srcp;
        float scale;
        if (half == 0) {
            float c = ok ? cnt[node] : 1.f;
            scale = ok ? (1.f / fmaxf(c, 1.f)) : 0.f;
            srcp = out + (ok ? node : 0) * 64;
        } else {
            scale = ok ? 1.f : 0.f;
            srcp = hid + (ok ? node : 0) * 64;
        }
        unsigned short* lrow = lds + row * 128 + half * 64;
        #pragma unroll
        for (int i = 0; i < 16; ++i) {
            v4f f = *(const v4f*)(srcp + 4 * i) * scale;
            v2u d;
            d.x = (unsigned)f32_to_bf16(f.x) | ((unsigned)f32_to_bf16(f.y) << 16);
            d.y = (unsigned)f32_to_bf16(f.z) | ((unsigned)f32_to_bf16(f.w) << 16);
            *(v2u*)(lrow + 8 * i) = d;
        }
    }
    __syncthreads();

    const int lane = tid & 31, wave = tid >> 5;
    v8f acc[4] = {v8f{}, v8f{}, v8f{}, v8f{}};
    wmma_tile(lds + (wave * 16 + (lane & 15)) * 128, w2p, lane, acc);

    const int  mbase = (lane >= 16) ? 8 : 0;
    const long nbase = base + wave * 16;
    #pragma unroll
    for (int n = 0; n < 4; ++n) {
        const int col = (n << 4) + (lane & 15);
        const float bias = b2[col];
        #pragma unroll
        for (int r = 0; r < 8; ++r) {
            long node = nbase + mbase + r;
            if (node < N) {
                float v = acc[n][r] + bias;
                out[node * 64 + col] = v > 0.f ? v : 0.f;
            }
        }
    }
}

// ---------------------------------------------------------------------------
extern "C" void kernel_launch(void* const* d_in, const int* in_sizes, int n_in,
                              void* d_out, int out_size, void* d_ws, size_t ws_size,
                              hipStream_t stream) {
    const float* hid = (const float*)d_in[0];
    const float* ef  = (const float*)d_in[1];
    const int*   src = (const int*)  d_in[2];
    const int*   dst = (const int*)  d_in[3];
    const float* W1  = (const float*)d_in[4];
    const float* b1  = (const float*)d_in[5];
    const float* W2  = (const float*)d_in[6];
    const float* b2  = (const float*)d_in[7];

    const int N = in_sizes[0] / 64;
    const int E = in_sizes[2];

    float* out = (float*)d_out;                     // doubles as msg_sum accumulator
    float* cnt = (float*)d_ws;
    size_t off = (((size_t)N * 4) + 255) & ~(size_t)255;
    unsigned short* w1p = (unsigned short*)((char*)d_ws + off);
    unsigned short* w2p = w1p + 8192;

    init_zero<<<2048, 256, 0, stream>>>(out, cnt, (long)N * 64, (long)N);
    pack_w<<<32, 256, 0, stream>>>(W1, w1p);
    pack_w<<<32, 256, 0, stream>>>(W2, w2p);
    edge_kernel<<<(E + 127) / 128, 256, 0, stream>>>(hid, ef, src, dst, w1p, b1, out, cnt, E);
    node_kernel<<<(N + 127) / 128, 256, 0, stream>>>(hid, w2p, b2, cnt, out, N);
}
